// MLP_Native_36129264894175
// MI455X (gfx1250) — matrix operand order repaired and checked by reading, not compiled, not run on hardware
//
#include <hip/hip_runtime.h>

typedef __attribute__((ext_vector_type(16))) _Float16 v16h;
typedef __attribute__((ext_vector_type(8)))  float    v8f;
typedef __attribute__((ext_vector_type(2)))  float    v2f;
typedef __attribute__((ext_vector_type(8)))  unsigned v8u;

#define N_POINTS (1u << 21)
#define TILES    ((int)(N_POINTS / 16))

// Packed f16 ReLU: single v_pk_max_num_f16, avoids LLVM's canonicalize+max expansion.
__device__ __forceinline__ unsigned relu_pk(unsigned v) {
    unsigned r;
    asm("v_pk_max_num_f16 %0, %1, 0" : "=v"(r) : "v"(v));
    return r;
}

__global__ __launch_bounds__(256)
void mlp_fused_wmma(const float* __restrict__ x,
                    const float* __restrict__ Wf,   // [64,3]
                    const float* __restrict__ W0,   // [64,64]
                    const float* __restrict__ W1,   // [64,64]
                    const float* __restrict__ Wl,   // [16,64]
                    float* __restrict__ out)        // [N]
{
    const int lane = threadIdx.x & 31;
    const int hi16 = lane >> 4;        // 0: lanes 0-15, 1: lanes 16-31
    const int l16  = lane & 15;
    const int wavesPerBlock = blockDim.x >> 5;
    const int wid = blockIdx.x * wavesPerBlock + (threadIdx.x >> 5);
    const int nw  = gridDim.x * wavesPerBlock;

    // ---- register-resident weight fragments (built once per wave) --------
    // f16 A 16x32 layout: half i of lane l holds k = k0 + (i&7) + (i&8?16:0) + (l>=16?8:0)
    // For fixed (row, c, hi16): k covers [base, base+7] and [base+16, base+23], base 32B-aligned.
    v16h a0[4][2], a1[4][2], al[2];
#pragma unroll
    for (int mt = 0; mt < 4; ++mt) {
      const int row = mt * 16 + l16;
#pragma unroll
      for (int c = 0; c < 2; ++c) {
        const int base = row * 64 + 32 * c + hi16 * 8;
        const float4* w0v = (const float4*)(W0 + base);
        const float4* w1v = (const float4*)(W1 + base);
        float t0[16], t1[16];
        *(float4*)(t0 + 0)  = w0v[0]; *(float4*)(t0 + 4)  = w0v[1];
        *(float4*)(t0 + 8)  = w0v[4]; *(float4*)(t0 + 12) = w0v[5];
        *(float4*)(t1 + 0)  = w1v[0]; *(float4*)(t1 + 4)  = w1v[1];
        *(float4*)(t1 + 8)  = w1v[4]; *(float4*)(t1 + 12) = w1v[5];
#pragma unroll
        for (int i = 0; i < 16; ++i) {
          a0[mt][c][i] = (_Float16)t0[i];
          a1[mt][c][i] = (_Float16)t1[i];
        }
      }
    }
#pragma unroll
    for (int c = 0; c < 2; ++c) {
      const int base = l16 * 64 + 32 * c + hi16 * 8;
      const float4* wlv = (const float4*)(Wl + base);
      float tl[16];
      *(float4*)(tl + 0)  = wlv[0]; *(float4*)(tl + 4)  = wlv[1];
      *(float4*)(tl + 8)  = wlv[4]; *(float4*)(tl + 12) = wlv[5];
#pragma unroll
      for (int i = 0; i < 16; ++i) al[c][i] = (_Float16)tl[i];
    }
    // f32 A 16x4 layout: vgpr0 holds K=0 (lanes<16) / K=2 (lanes>=16); vgpr1 K=1 / K=3(pad 0)
    v2f af[4];
#pragma unroll
    for (int mt = 0; mt < 4; ++mt) {
      const int row = mt * 16 + l16;
      af[mt].x = Wf[row * 3 + (hi16 ? 2 : 0)];
      af[mt].y = hi16 ? 0.f : Wf[row * 3 + 1];
    }

    const v8f zero8 = {};

    // D (f32, [neuron][point]) tiles -> next layer's f16 B fragment.
    // No lane swap: half h gets rows 8h..8h+7 of tile 2c (elems 0-7) and of tile 2c+1 (elems 8-15).
    // ReLU is applied packed (v_pk_max_num_f16) after the f16 pack.
    auto buildB = [&](const v8f th[4], v16h B[2]) {
#pragma unroll
      for (int c = 0; c < 2; ++c) {
        float lo[8], hi[8];
#pragma unroll
        for (int i = 0; i < 8; ++i) {
          lo[i] = th[2 * c][i];
          hi[i] = th[2 * c + 1][i];
        }
        v8u bb;
#pragma unroll
        for (int j = 0; j < 4; ++j) {
          bb[j]     = relu_pk(__builtin_bit_cast(unsigned,
                          __builtin_amdgcn_cvt_pkrtz(lo[2 * j], lo[2 * j + 1])));
          bb[4 + j] = relu_pk(__builtin_bit_cast(unsigned,
                          __builtin_amdgcn_cvt_pkrtz(hi[2 * j], hi[2 * j + 1])));
        }
        B[c] = __builtin_bit_cast(v16h, bb);
      }
    };

    // ---- software-pipelined tile loop ------------------------------------
    int t = wid;
    if (t >= TILES) return;
    int pc = t * 16 + l16;                       // current tile point
    float xa = x[pc * 3 + (hi16 ? 2 : 0)];
    float xb = x[pc * 3 + 1];

    while (t < TILES) {
      const int tn = t + nw;
      const int tcl = (tn < TILES) ? tn : (TILES - 1);   // clamped, uniform
      const int pn = tcl * 16 + l16;
      float nxa = x[pn * 3 + (hi16 ? 2 : 0)];            // prefetch next tile's x
      float nxb = x[pn * 3 + 1];

      // ---- layer 1 (f32 WMMA, K=4 padded from 3): D = Wf @ X^T ----------
      v2f bf;
      bf.x = xa;
      bf.y = hi16 ? 0.f : xb;
      v8f h1[4];
#pragma unroll
      for (int mt = 0; mt < 4; ++mt)
        h1[mt] = __builtin_amdgcn_wmma_f32_16x16x4_f32(
            false, af[mt], false, bf, (short)0, zero8, false, false);

      // ---- hidden layer 2 -----------------------------------------------
      v16h B[2];
      buildB(h1, B);
      v8f h2[4];
#pragma unroll
      for (int mt = 0; mt < 4; ++mt) {
        v8f acc = __builtin_amdgcn_wmma_f32_16x16x32_f16(
            false, a0[mt][0], false, B[0], (short)0, zero8, false, false);
        h2[mt] = __builtin_amdgcn_wmma_f32_16x16x32_f16(
            false, a0[mt][1], false, B[1], (short)0, acc, false, false);
      }

      // ---- hidden layer 3 -----------------------------------------------
      buildB(h2, B);
      v8f h3[4];
#pragma unroll
      for (int mt = 0; mt < 4; ++mt) {
        v8f acc = __builtin_amdgcn_wmma_f32_16x16x32_f16(
            false, a1[mt][0], false, B[0], (short)0, zero8, false, false);
        h3[mt] = __builtin_amdgcn_wmma_f32_16x16x32_f16(
            false, a1[mt][1], false, B[1], (short)0, acc, false, false);
      }

      // ---- last layer (16 output rows, keep row 0) ----------------------
      buildB(h3, B);
      v8f o = __builtin_amdgcn_wmma_f32_16x16x32_f16(
          false, al[0], false, B[0], (short)0, zero8, false, false);
      o = __builtin_amdgcn_wmma_f32_16x16x32_f16(
          false, al[1], false, B[1], (short)0, o, false, false);

      if (!hi16) out[t * 16 + l16] = o[0];  // D vgpr0, lanes 0-15 = (M=0, N=0..15)

      t = tn;
      xa = nxa;
      xb = nxb;
    }
}

extern "C" void kernel_launch(void* const* d_in, const int* in_sizes, int n_in,
                              void* d_out, int out_size, void* d_ws, size_t ws_size,
                              hipStream_t stream) {
    const float* x  = (const float*)d_in[0];
    const float* Wf = (const float*)d_in[1];
    const float* W0 = (const float*)d_in[2];
    const float* W1 = (const float*)d_in[3];
    const float* Wl = (const float*)d_in[4];
    float* out = (float*)d_out;

    dim3 grid(2048), block(256);   // 16384 waves -> 8 tiles (128 pts) per wave
    mlp_fused_wmma<<<grid, block, 0, stream>>>(x, Wf, W0, W1, Wl, out);
}
